// VLACpuRef_78056735637555
// MI455X (gfx1250) — matrix operand order repaired and checked by reading, not compiled, not run on hardware
//
#include <hip/hip_runtime.h>
#include <hip/hip_bf16.h>
#include <math.h>

typedef __attribute__((ext_vector_type(16))) _Float16 v16h;
typedef __attribute__((ext_vector_type(8)))  float    v8f;

#define ACT_NONE 0
#define ACT_GELU 1
#define ACT_SILU 2

// ---------------------------------------------------------------------------
// WMMA GEMM: C[M,N] = epilogue(A[M,K] x B) ; B is [K,N] row-major, or (TRANSB)
// stored [N,K] row-major. One wave32 per 16x16 tile, blockDim=(32,4): the 4
// waves of a block share tileN, so the B tile is staged once per block in LDS
// (f16, transposed [N][K], stride 40 halves = 80B so each lane's 16-half
// fragment is contiguous & 16B aligned), double-buffered with one barrier per
// K-step. All waves stay barrier-live; out-of-range M tiles compute on zero A.
// ---------------------------------------------------------------------------
template<bool TRANSB>
__global__ void __launch_bounds__(128)
gemm_wmma_kernel(const float* __restrict__ A, const float* __restrict__ B,
                 const float* __restrict__ bias, const float* __restrict__ residual,
                 float* __restrict__ C, int M, int N, int K, int act)
{
    __shared__ _Float16 Bs[2][16 * 40];

    const int lane  = threadIdx.x;                       // 0..31
    const int tileN = blockIdx.x;
    const int tileM = blockIdx.y * blockDim.y + threadIdx.y;
    const int ncol0 = tileN * 16;
    const int ncol  = ncol0 + (lane & 15);
    const int mrow  = tileM * 16 + (lane & 15);
    const int aKoff = (lane >> 4) ? 16 : 0;              // A frag K sub-offset
    const int bKoff = (lane >> 4) ? 16 : 0;              // B frag K sub-offset
    const bool arow_ok = (mrow < M);

    if (TRANSB) {                                        // no barriers below in
        if (tileM * 16 >= M) return;                     // this path: safe exit
    }

    // cooperative B-tile fill assignment: 128 threads x float4 = 32x16 floats
    const int tid = threadIdx.y * 32 + lane;
    const int fk  = tid >> 2;                            // 0..31 (K row)
    const int fn  = (tid & 3) * 4;                       // 0,4,8,12 (N quad)

    if (!TRANSB) {
        const float* gp = B + (size_t)fk * N + ncol0 + fn;
#pragma unroll
        for (int j = 0; j < 4; ++j)
            Bs[0][(fn + j) * 40 + fk] = (_Float16)gp[j];
    }

    v8f acc = {};
    for (int kb = 0; kb < K; kb += 32) {
        const int cur = (kb >> 5) & 1;

        if (!TRANSB) {
            __syncthreads();                             // fill(cur) visible;
            if (kb + 32 < K) {                           // prev reads retired
                const float* gp = B + (size_t)(kb + 32 + fk) * N + ncol0 + fn;
                __builtin_prefetch(gp + (size_t)32 * N, 0, 1);
#pragma unroll
                for (int j = 0; j < 4; ++j)
                    Bs[cur ^ 1][(fn + j) * 40 + fk] = (_Float16)gp[j];
            }
        }

        // A fragment (16x32 f16): halves 0..15 -> K=kb+aKoff+i
        v16h a;
        if (arow_ok) {
            const float* ap = A + (size_t)mrow * K + kb + aKoff;
#pragma unroll
            for (int i = 0; i < 8; ++i) a[i]     = (_Float16)ap[i];
#pragma unroll
            for (int i = 0; i < 8; ++i) a[8 + i] = (_Float16)ap[8 + i];
            __builtin_prefetch(ap + 32, 0, 1);
        } else {
#pragma unroll
            for (int i = 0; i < 16; ++i) a[i] = (_Float16)0.0f;
        }

        // B fragment (32x16 f16): lane holds column ncol, K=kb+bKoff+{0..15}
        v16h b;
        if (TRANSB) {
            const float* bp = B + (size_t)ncol * K + kb + bKoff;
#pragma unroll
            for (int i = 0; i < 16; ++i) b[i] = (_Float16)bp[i];
        } else {
            const _Float16* bp = &Bs[cur][(lane & 15) * 40 + bKoff];
#pragma unroll
            for (int i = 0; i < 16; ++i) b[i] = bp[i];   // 2x ds_load_b128
        }

        acc = __builtin_amdgcn_wmma_f32_16x16x32_f16(false, a, false, b,
                                                     (short)0, acc, false, false);
    }

    // Epilogue: lanes 0..15 own rows tileM*16+{0..7}; lanes 16..31 rows +8.
    const int mbase = tileM * 16 + ((lane >> 4) ? 8 : 0);
    const float bv = bias ? bias[ncol] : 0.0f;
#pragma unroll
    for (int i = 0; i < 8; ++i) {
        int m = mbase + i;
        if (m < M) {
            float v = acc[i] + bv;
            if (act == ACT_GELU)      v = 0.5f * v * (1.0f + erff(v * 0.70710678118f));
            else if (act == ACT_SILU) v = v / (1.0f + __expf(-v));
            if (residual) v += residual[(size_t)m * N + ncol];
            C[(size_t)m * N + ncol] = v;
        }
    }
}

// ---------------------------------------------------------------------------
// Online-softmax attention, head_dim = 64, GQA (q head h uses kv head h/rep).
// One thread per (query, q-head); q and acc held fully in registers.
// ---------------------------------------------------------------------------
__global__ void attn_kernel(const float* __restrict__ Q, int sq,
                            const float* __restrict__ K, int sk,
                            const float* __restrict__ V, int sv,
                            float* __restrict__ O, int so,
                            int Lq, int Lk, int QH, int KVH, int causal)
{
    int t = blockIdx.x * blockDim.x + threadIdx.x;
    if (t >= Lq * QH) return;
    int q = t / QH, h = t % QH;
    int kvh = h / (QH / KVH);

    const float* qp = Q + (size_t)q * sq + h * 64;
    float qr[64];
#pragma unroll
    for (int d = 0; d < 64; ++d) qr[d] = qp[d];

    float m = -3.0e38f, l = 0.0f, acc[64];
#pragma unroll
    for (int d = 0; d < 64; ++d) acc[d] = 0.0f;

    int kend = causal ? (q + 1) : Lk;
    for (int k = 0; k < kend; ++k) {
        const float* kp = K + (size_t)k * sk + kvh * 64;
        float s = 0.0f;
#pragma unroll
        for (int d = 0; d < 64; ++d) s += qr[d] * kp[d];
        s *= 0.125f;
        float mn   = fmaxf(m, s);
        float corr = __expf(m - mn);
        float w    = __expf(s - mn);
        const float* vp = V + (size_t)k * sv + kvh * 64;
        l = l * corr + w;
#pragma unroll
        for (int d = 0; d < 64; ++d) acc[d] = acc[d] * corr + w * vp[d];
        m = mn;
    }
    float inv = 1.0f / l;
    float* op = O + (size_t)q * so + h * 64;
#pragma unroll
    for (int d = 0; d < 64; ++d) op[d] = acc[d] * inv;
}

__global__ void rope_kernel(float* __restrict__ X, int L, int H, int stride)
{
    int t = blockIdx.x * blockDim.x + threadIdx.x;
    if (t >= L * H * 32) return;
    int d = t & 31;
    int h = (t >> 5) % H;
    int l = t / (32 * H);
    float rad = (float)l * __powf(10000.0f, -(float)d * (1.0f / 32.0f));
    float s = __sinf(rad), c = __cosf(rad);
    float* p = X + (size_t)l * stride + h * 64 + d;
    float x1 = p[0], x2 = p[32];
    p[0]  = x1 * c - x2 * s;
    p[32] = x2 * c + x1 * s;
}

__global__ void rmsnorm_kernel(const float* __restrict__ X, const float* __restrict__ g,
                               float* __restrict__ Y, int N)
{
    __shared__ float red[256];
    int row = blockIdx.x;
    const float* x = X + (size_t)row * N;
    float s = 0.0f;
    for (int i = threadIdx.x; i < N; i += 256) { float v = x[i]; s += v * v; }
    red[threadIdx.x] = s; __syncthreads();
    for (int o = 128; o > 0; o >>= 1) {
        if ((int)threadIdx.x < o) red[threadIdx.x] += red[threadIdx.x + o];
        __syncthreads();
    }
    float inv = rsqrtf(red[0] / (float)N + 1.1920929e-07f);
    for (int i = threadIdx.x; i < N; i += 256)
        Y[(size_t)row * N + i] = x[i] * inv * g[i];
}

__global__ void layernorm_kernel(const float* __restrict__ X, const float* __restrict__ g,
                                 const float* __restrict__ b, float* __restrict__ Y, int N)
{
    __shared__ float r1[256], r2[256];
    int row = blockIdx.x;
    const float* x = X + (size_t)row * N;
    float s1 = 0.0f, s2 = 0.0f;
    for (int i = threadIdx.x; i < N; i += 256) { float v = x[i]; s1 += v; s2 += v * v; }
    r1[threadIdx.x] = s1; r2[threadIdx.x] = s2; __syncthreads();
    for (int o = 128; o > 0; o >>= 1) {
        if ((int)threadIdx.x < o) {
            r1[threadIdx.x] += r1[threadIdx.x + o];
            r2[threadIdx.x] += r2[threadIdx.x + o];
        }
        __syncthreads();
    }
    float mean = r1[0] / (float)N;
    float var  = r2[0] / (float)N - mean * mean;
    float inv  = rsqrtf(var + 1e-5f);
    for (int i = threadIdx.x; i < N; i += 256)
        Y[(size_t)row * N + i] = (x[i] - mean) * inv * g[i] + b[i];
}

__global__ void mul_kernel(const float* a, const float* b, float* o, int n)
{ int i = blockIdx.x * blockDim.x + threadIdx.x; if (i < n) o[i] = a[i] * b[i]; }

__global__ void copy_kernel(const float* a, float* o, int n)
{ int i = blockIdx.x * blockDim.x + threadIdx.x; if (i < n) o[i] = a[i]; }

__global__ void fill0_kernel(float* o, int n)
{ int i = blockIdx.x * blockDim.x + threadIdx.x; if (i < n) o[i] = 0.0f; }

__global__ void im2col_kernel(const float* __restrict__ img, float* __restrict__ A)
{
    int t = blockIdx.x * blockDim.x + threadIdx.x;
    if (t >= 1024 * 768) return;
    int p = t / 768, j = t % 768;
    int c = j / 256, r = (j % 256) / 16, cc = j % 16;
    int ph = p / 32, pw = p % 32;
    A[t] = img[(size_t)c * 512 * 512 + (size_t)(ph * 16 + r) * 512 + pw * 16 + cc];
}

__global__ void connector_gather_kernel(const float* __restrict__ xv, float* __restrict__ V)
{
    int t = blockIdx.x * blockDim.x + threadIdx.x;
    if (t >= 64 * 12288) return;
    int tok = t / 12288, f = t % 12288;
    int h1 = tok / 8, w1 = tok % 8;
    int dh = f / 3072, dw = (f % 3072) / 768, e = f % 768;
    int p = (h1 * 4 + dh) * 32 + (w1 * 4 + dw);
    V[t] = xv[(size_t)p * 768 + e];
}

static void gemm(hipStream_t s, const float* A, const float* B, const float* bias,
                 const float* res, float* C, int M, int N, int K, int act, bool transB)
{
    dim3 grid(N / 16, (((M + 15) / 16) + 3) / 4);
    dim3 blk(32, 4);
    if (transB) gemm_wmma_kernel<true ><<<grid, blk, 0, s>>>(A, B, bias, res, C, M, N, K, act);
    else        gemm_wmma_kernel<false><<<grid, blk, 0, s>>>(A, B, bias, res, C, M, N, K, act);
}
static inline int cdiv(int a, int b) { return (a + b - 1) / b; }

// Input indices: JAX pytree flattening (dict keys sorted at each level).
enum {
    IN_ACT = 0, IN_IMG = 1,
    EC_DOWN = 2, EC_GATE, EC_K, EC_LN1, EC_LN2, EC_O, EC_Q, EC_ROPE, EC_UP, EC_V,
    ES_DOWN = 12, ES_GATE, ES_K, ES_LN1, ES_LN2, ES_O, ES_Q, ES_ROPE, ES_UP, ES_V,
    M_CONN = 22, M_CONVB, M_CONVW, M_POSTLN, M_POSTW, M_STATEW, M_TEXTEMB,
    TX_DOWN = 29, TX_GATE, TX_K, TX_LN1, TX_LN2, TX_O, TX_Q, TX_ROPE, TX_UP, TX_V,
    V_FFND = 39, V_FFNU, V_INPROJ, V_LN1B, V_LN1G, V_LN2B, V_LN2G, V_OUTPROJ,
    IN_STATE = 47
};

extern "C" void kernel_launch(void* const* d_in, const int* in_sizes, int n_in,
                              void* d_out, int out_size, void* d_ws, size_t ws_size,
                              hipStream_t stream)
{
    (void)in_sizes; (void)n_in; (void)out_size; (void)ws_size;
    hipStream_t s = stream;
    #define FP(i) ((const float*)d_in[i])

    char* wp = (char*)d_ws;
    auto alloc = [&](size_t nf) -> float* {
        float* p = (float*)wp; wp += ((nf * 4 + 255) / 256) * 256; return p;
    };
    float* aimg = alloc(1024 * 768);
    float* xv   = alloc(1024 * 768);
    float* hv   = alloc(1024 * 768);
    float* qkv  = alloc(1024 * 2304);
    float* ctxv = alloc(1024 * 768);
    float* ffnv = alloc(1024 * 3072);
    float* vtok = alloc(64 * 12288);
    float* xt   = alloc(256 * 960);
    float* ht   = alloc(256 * 960);
    float* qt   = alloc(256 * 960);
    float* kt   = alloc(256 * 320);
    float* vt   = alloc(256 * 320);
    float* ctxt = alloc(256 * 960);
    float* gat  = alloc(256 * 2560);
    float* upt  = alloc(256 * 2560);
    float* tk   = alloc(256 * 320);
    float* tv   = alloc(256 * 320);
    float* kx   = alloc(256 * 320);
    float* vx   = alloc(256 * 320);
    float* xa   = alloc(64 * 960);
    float* ha   = alloc(64 * 960);
    float* qa   = alloc(64 * 960);
    float* ka   = alloc(64 * 320);
    float* va   = alloc(64 * 320);
    float* ctxa = alloc(64 * 960);
    float* gaa  = alloc(64 * 2560);
    float* upa  = alloc(64 * 2560);

    #define EW(kern, n, ...) kern<<<cdiv((n), 256), 256, 0, s>>>(__VA_ARGS__)

    EW(im2col_kernel, 1024 * 768, FP(IN_IMG), aimg);
    gemm(s, aimg, FP(M_CONVW), FP(M_CONVB), nullptr, xv, 1024, 768, 768, ACT_NONE, true);

    for (int it = 0; it < 12; ++it) {
        layernorm_kernel<<<1024, 256, 0, s>>>(xv, FP(V_LN1G), FP(V_LN1B), hv, 768);
        gemm(s, hv, FP(V_INPROJ), nullptr, nullptr, qkv, 1024, 2304, 768, ACT_NONE, false);
        attn_kernel<<<cdiv(1024 * 12, 128), 128, 0, s>>>(
            qkv + 0, 2304, qkv + 768, 2304, qkv + 1536, 2304,
            ctxv, 768, 1024, 1024, 12, 12, 0);
        gemm(s, ctxv, FP(V_OUTPROJ), nullptr, xv, xv, 1024, 768, 768, ACT_NONE, false);
        layernorm_kernel<<<1024, 256, 0, s>>>(xv, FP(V_LN2G), FP(V_LN2B), hv, 768);
        gemm(s, hv, FP(V_FFNU), nullptr, nullptr, ffnv, 1024, 3072, 768, ACT_GELU, false);
        gemm(s, ffnv, FP(V_FFND), nullptr, xv, xv, 1024, 768, 3072, ACT_NONE, false);
    }

    EW(connector_gather_kernel, 64 * 12288, xv, vtok);
    gemm(s, vtok, FP(M_CONN), nullptr, nullptr, xt, 64, 960, 12288, ACT_NONE, false);
    EW(copy_kernel, 128 * 960, FP(M_TEXTEMB), xt + (size_t)64 * 960, 128 * 960);
    gemm(s, FP(IN_STATE), FP(M_STATEW), nullptr, nullptr, xt + (size_t)192 * 960,
         1, 960, 32, ACT_NONE, false);
    EW(fill0_kernel, 63 * 960, xt + (size_t)193 * 960, 63 * 960);
    EW(copy_kernel, 50 * 960, FP(IN_ACT), xa, 50 * 960);

    for (int i = 0; i < 16; ++i) {
        rmsnorm_kernel<<<256, 256, 0, s>>>(xt, FP(TX_LN1), ht, 960);
        gemm(s, ht, FP(TX_Q), nullptr, nullptr, qt, 256, 960, 960, ACT_NONE, false);
        gemm(s, ht, FP(TX_K), nullptr, nullptr, kt, 256, 320, 960, ACT_NONE, false);
        gemm(s, ht, FP(TX_V), nullptr, nullptr, vt, 256, 320, 960, ACT_NONE, false);
        EW(rope_kernel, 256 * 15 * 32, qt, 256, 15, 960);
        EW(rope_kernel, 256 * 5 * 32,  kt, 256, 5, 320);
        attn_kernel<<<cdiv(256 * 15, 128), 128, 0, s>>>(qt, 960, kt, 320, vt, 320,
                                                        ctxt, 960, 256, 256, 15, 5, 1);
        gemm(s, ctxt, FP(TX_O), nullptr, xt, xt, 256, 960, 960, ACT_NONE, false);
        rmsnorm_kernel<<<256, 256, 0, s>>>(xt, FP(TX_LN2), ht, 960);
        gemm(s, ht, FP(TX_GATE), nullptr, nullptr, gat, 256, 2560, 960, ACT_SILU, false);
        gemm(s, ht, FP(TX_UP),   nullptr, nullptr, upt, 256, 2560, 960, ACT_NONE, false);
        EW(mul_kernel, 256 * 2560, gat, upt, gat, 256 * 2560);
        gemm(s, gat, FP(TX_DOWN), nullptr, xt, xt, 256, 960, 2560, ACT_NONE, false);

        if ((i & 1) == 0) {
            rmsnorm_kernel<<<50, 256, 0, s>>>(xa, FP(ES_LN1), ha, 960);
            gemm(s, ha, FP(ES_Q), nullptr, nullptr, qa, 50, 960, 960, ACT_NONE, false);
            gemm(s, ha, FP(ES_K), nullptr, nullptr, ka, 50, 320, 960, ACT_NONE, false);
            gemm(s, ha, FP(ES_V), nullptr, nullptr, va, 50, 320, 960, ACT_NONE, false);
            EW(rope_kernel, 50 * 15 * 32, qa, 50, 15, 960);
            EW(rope_kernel, 50 * 5 * 32,  ka, 50, 5, 320);
            attn_kernel<<<cdiv(50 * 15, 128), 128, 0, s>>>(qa, 960, ka, 320, va, 320,
                                                           ctxa, 960, 50, 50, 15, 5, 1);
            gemm(s, ctxa, FP(ES_O), nullptr, xa, xa, 50, 960, 960, ACT_NONE, false);
            rmsnorm_kernel<<<50, 256, 0, s>>>(xa, FP(ES_LN2), ha, 960);
            gemm(s, ha, FP(ES_GATE), nullptr, nullptr, gaa, 50, 2560, 960, ACT_SILU, false);
            gemm(s, ha, FP(ES_UP),   nullptr, nullptr, upa, 50, 2560, 960, ACT_NONE, false);
            EW(mul_kernel, 50 * 2560, gaa, upa, gaa, 50 * 2560);
            gemm(s, gaa, FP(ES_DOWN), nullptr, xa, xa, 50, 960, 2560, ACT_NONE, false);
        } else {
            rmsnorm_kernel<<<256, 256, 0, s>>>(xt, FP(TX_LN1), ht, 960);
            gemm(s, ht, FP(TX_K), nullptr, nullptr, tk, 256, 320, 960, ACT_NONE, false);
            gemm(s, ht, FP(TX_V), nullptr, nullptr, tv, 256, 320, 960, ACT_NONE, false);
            rmsnorm_kernel<<<50, 256, 0, s>>>(xa, FP(EC_LN1), ha, 960);
            gemm(s, ha, FP(EC_Q), nullptr, nullptr, qa, 50, 960, 960, ACT_NONE, false);
            EW(rope_kernel, 50 * 15 * 32, qa, 50, 15, 960);
            gemm(s, tk, FP(EC_K), nullptr, nullptr, kx, 256, 320, 320, ACT_NONE, false);
            gemm(s, tv, FP(EC_V), nullptr, nullptr, vx, 256, 320, 320, ACT_NONE, false);
            attn_kernel<<<cdiv(50 * 15, 128), 128, 0, s>>>(qa, 960, kx, 320, vx, 320,
                                                           ctxa, 960, 50, 256, 15, 5, 0);
            gemm(s, ctxa, FP(EC_O), nullptr, xa, xa, 50, 960, 960, ACT_NONE, false);
            rmsnorm_kernel<<<50, 256, 0, s>>>(xa, FP(EC_LN2), ha, 960);
            gemm(s, ha, FP(EC_GATE), nullptr, nullptr, gaa, 50, 2560, 960, ACT_SILU, false);
            gemm(s, ha, FP(EC_UP),   nullptr, nullptr, upa, 50, 2560, 960, ACT_NONE, false);
            EW(mul_kernel, 50 * 2560, gaa, upa, gaa, 50 * 2560);
            gemm(s, gaa, FP(EC_DOWN), nullptr, xa, xa, 50, 960, 2560, ACT_NONE, false);
        }
    }

    rmsnorm_kernel<<<50, 256, 0, s>>>(xa, FP(M_POSTLN), ha, 960);
    gemm(s, ha, FP(M_POSTW), nullptr, nullptr, (float*)d_out, 50, 32, 960, ACT_NONE, false);
}
